// SchNetModel_33208687133092
// MI455X (gfx1250) — matrix operand order repaired and checked by reading, not compiled, not run on hardware
//
#include <hip/hip_runtime.h>
#include <hip/hip_bf16.h>

typedef __attribute__((ext_vector_type(16))) __bf16 v16bf;
typedef __attribute__((ext_vector_type(8)))  float  v8f;

#define HD 128
#define FD 64
#define NGAUSS 50
#define PI_F 3.14159265358979f
#define CUT 5.0f
#define WPB 4                 // waves per block in edge kernels
#define LAYER_WSHORTS 32768   // bf16 weight table shorts per layer

// ---------- scalar helpers ----------
__device__ __forceinline__ float sspf(float x) {        // softplus(x) - log(2)
    return fmaxf(x, 0.f) + log1pf(expf(-fabsf(x))) - 0.6931471805599453f;
}
__device__ __forceinline__ float sigf(float x) { return 1.f / (1.f + expf(-x)); }

__device__ __forceinline__ unsigned short bfbits(float f) {
    __bf16 h = (__bf16)f; unsigned short u; __builtin_memcpy(&u, &h, 2); return u;
}
__device__ __forceinline__ __bf16 bffrom(unsigned short u) {
    __bf16 h; __builtin_memcpy(&h, &u, 2); return h;
}
__device__ __forceinline__ unsigned int pkbf(float lo, float hi) {
    return (unsigned int)bfbits(lo) | ((unsigned int)bfbits(hi) << 16);
}

__device__ __forceinline__ v8f wmma_bf16(v16bf a, v16bf b, v8f c) {
    // D = A(16x32 bf16) * B(32x16 bf16) + C(16x16 f32)
    return __builtin_amdgcn_wmma_f32_16x16x32_bf16(false, a, false, b, (short)0, c, false, false);
}

// A-fragment K index base: lanes0-15 M=lane, lanes16-31 M=lane-16 (other K half)
// VGPR v: k = ((v&4)?16:0) + half*8 + (v&3)*2 (+p)
__device__ __forceinline__ int akbase(int v, int half) {
    return ((v & 4) ? 16 : 0) + half * 8 + (v & 3) * 2;
}

union VB { v16bf v; unsigned int u[8]; };

// Load A fragment from an LDS [16][stride] bf16 matrix (row-major, m = lane&15)
__device__ __forceinline__ v16bf loadA_lds(const unsigned short* arr, int stride,
                                           int lm, int half, int koff) {
    VB r;
#pragma unroll
    for (int v = 0; v < 8; ++v) {
        int k = koff + akbase(v, half);
        r.u[v] = *(const unsigned int*)(arr + lm * stride + k);
    }
    return r.v;
}

// Load B fragment from bf16 weight table stored as Bt[n][Kpad] with Bt[n][k]=B[k][n]
// VGPR v: k = ((v&4)?16:0) + half*8 + (v&3)*2 (+p) (same as A); n = lane&15 (+16*nt)
__device__ __forceinline__ v16bf loadB(const unsigned short* wt, int Kpad, int n,
                                       int half, int koff) {
    VB r;
#pragma unroll
    for (int v = 0; v < 8; ++v) {
        int k = koff + akbase(v, half);
        r.u[v] = *(const unsigned int*)(wt + n * Kpad + k);
    }
    return r.v;
}

// Load A fragment from a global float row (h[j] row), convert to bf16
__device__ __forceinline__ v16bf loadA_rowf(const float* row, int koff, int half) {
    VB r;
#pragma unroll
    for (int v = 0; v < 8; ++v) {
        int k = koff + akbase(v, half);
        float2 f = *(const float2*)(row + k);
        r.u[v] = pkbf(f.x, f.y);
    }
    return r.v;
}

// ---------- small utility kernels ----------
__global__ void zero_kernel(float* p, long n) {
    long i = (long)blockIdx.x * blockDim.x + threadIdx.x;
    long stride = (long)gridDim.x * blockDim.x;
    for (; i < n; i += stride) p[i] = 0.f;
}

__global__ void geom_kernel(const float* ew, float* dvec, float* dd, int E) {
    int e = blockIdx.x * blockDim.x + threadIdx.x;
    if (e >= E) return;
    float a = ew[3 * e], b = ew[3 * e + 1], c = ew[3 * e + 2];
    dvec[e] = sqrtf(a * a + b * b + c * c);
    dd[e] = 0.f;
}

__global__ void embed_kernel(const int* x, const float* emb, float* h0, int N) {
    int idx = blockIdx.x * blockDim.x + threadIdx.x;
    if (idx >= N * HD) return;
    int n = idx / HD, c = idx % HD;
    h0[idx] = emb[x[n] * HD + c];
}

// Convert layer weights to bf16 tables (each [n][Kpad], Bt[n][k]=B[k][n]).
// Regions (shorts): B1F 0..4095, B2F 4096..8191, B3F 8192..16383,
//                   B2B 16384..20479, B1B 20480..24575, B3B 24576..32767
__global__ void prep_kernel(const float* w1, const float* w2, const float* lin1,
                            unsigned short* wb) {
    int idx = blockIdx.x * blockDim.x + threadIdx.x;
    if (idx >= LAYER_WSHORTS) return;
    unsigned short out;
    if (idx < 4096) {                    // B1F[f][ng] = W1[ng][f], ng<50
        int f = idx / 64, ng = idx % 64;
        out = bfbits((ng < NGAUSS) ? w1[ng * 64 + f] : 0.f);
    } else if (idx < 8192) {             // B2F[n][k] = W2[k][n]
        int r = idx - 4096; int n = r / 64, k = r % 64;
        out = bfbits(w2[k * 64 + n]);
    } else if (idx < 16384) {            // B3F[n=64][k=128] = lin1[k][n]
        int r = idx - 8192; int n = r / 128, k = r % 128;
        out = bfbits(lin1[k * 64 + n]);
    } else if (idx < 20480) {            // B2B[n][k] = W2[n][k]
        int r = idx - 16384;
        out = bfbits(w2[r]);
    } else if (idx < 24576) {            // B1B[n][k] = W1[n][k], n<50
        int r = idx - 20480; int n = r / 64;
        out = (n < NGAUSS) ? bfbits(w1[r]) : (unsigned short)0;
    } else {                             // B3B[n=128][k=64] = lin1[n][k]
        int r = idx - 24576;
        out = bfbits(lin1[r]);
    }
    wb[idx] = out;
}

// ---------- forward edge kernel (WMMA) ----------
__global__ void edge_fwd_kernel(const int* ei, const float* dvec, const float* h_in,
                                float* agg, const unsigned short* wb,
                                const float* b1, const float* b2, int E, int tiles) {
    __shared__ __align__(16) unsigned short rbfs[WPB][16 * 64];
    __shared__ __align__(16) unsigned short a1s[WPB][16 * 64];
    __shared__ float dtile[WPB][16];
    __shared__ float Ctile[WPB][16];
    __shared__ int   itile[WPB][16];
    __shared__ int   jtile[WPB][16];

    int wave = threadIdx.x >> 5, lane = threadIdx.x & 31;
    int lm = lane & 15, half = lane >> 4;
    int tile = blockIdx.x * WPB + wave;
    bool valid = tile < tiles;
    int tbase = valid ? tile * 16 : 0;

    if (lane < 16) {
        int e = tbase + lane; if (e >= E) e = E - 1;
        float d = dvec[e];
        dtile[wave][lane] = d;
        Ctile[wave][lane] = (d <= CUT) ? 0.5f * (cosf(d * PI_F / CUT) + 1.f) : 0.f;
        itile[wave][lane] = ei[e];
        jtile[wave][lane] = ei[E + e];
    }
    __syncthreads();

    // precompute rbf tile once: 16 edges x 64 (k>=50 zero-padded)
    for (int idx = lane; idx < 16 * 64; idx += 32) {
        int m = idx >> 6, k = idx & 63;
        float val = 0.f;
        if (k < NGAUSS) {
            float t = dtile[wave][m] - (CUT / (NGAUSS - 1)) * (float)k;
            val = expf(-50.f * t * t);   // coeff = -0.5/width^2 = -50
        }
        rbfs[wave][idx] = bfbits(val);
    }
    __syncthreads();

    const unsigned short* B1F = wb;
    const unsigned short* B2F = wb + 4096;
    const unsigned short* B3F = wb + 8192;

    // GEMM1: t1 = rbf @ W1 + b1 ; a1 = ssp(t1) -> LDS (bf16)
    {
        v16bf A0 = loadA_lds(rbfs[wave], 64, lm, half, 0);
        v16bf A1 = loadA_lds(rbfs[wave], 64, lm, half, 32);
#pragma unroll
        for (int nt = 0; nt < 4; ++nt) {
            float bias = b1[nt * 16 + lm];
            v8f acc;
#pragma unroll
            for (int q = 0; q < 8; ++q) acc[q] = bias;
            acc = wmma_bf16(A0, loadB(B1F, 64, nt * 16 + lm, half, 0), acc);
            acc = wmma_bf16(A1, loadB(B1F, 64, nt * 16 + lm, half, 32), acc);
#pragma unroll
            for (int q = 0; q < 8; ++q) {
                int m = q + 8 * half;
                a1s[wave][m * 64 + nt * 16 + lm] = bfbits(sspf(acc[q]));
            }
        }
    }
    __syncthreads();

    // GEMM2: t2 = a1 @ W2 + b2 ; Wf = t2 * C
    v8f Wf[4];
    {
        v16bf A0 = loadA_lds(a1s[wave], 64, lm, half, 0);
        v16bf A1 = loadA_lds(a1s[wave], 64, lm, half, 32);
#pragma unroll
        for (int nt = 0; nt < 4; ++nt) {
            float bias = b2[nt * 16 + lm];
            v8f acc;
#pragma unroll
            for (int q = 0; q < 8; ++q) acc[q] = bias;
            acc = wmma_bf16(A0, loadB(B2F, 64, nt * 16 + lm, half, 0), acc);
            acc = wmma_bf16(A1, loadB(B2F, 64, nt * 16 + lm, half, 32), acc);
#pragma unroll
            for (int q = 0; q < 8; ++q) Wf[nt][q] = acc[q] * Ctile[wave][q + 8 * half];
        }
    }

    // GEMM3: g = h[j] @ lin1_w ; msg = Wf*g ; scatter agg[i] += msg
    {
        const float* hrow = h_in + (size_t)jtile[wave][lm] * HD;
        v16bf HA[4];
#pragma unroll
        for (int kk = 0; kk < 4; ++kk) HA[kk] = loadA_rowf(hrow, kk * 32, half);
#pragma unroll
        for (int nt = 0; nt < 4; ++nt) {
            v8f acc;
#pragma unroll
            for (int q = 0; q < 8; ++q) acc[q] = 0.f;
#pragma unroll
            for (int kk = 0; kk < 4; ++kk)
                acc = wmma_bf16(HA[kk], loadB(B3F, 128, nt * 16 + lm, half, kk * 32), acc);
            if (valid) {
#pragma unroll
                for (int q = 0; q < 8; ++q) {
                    int m = q + 8 * half;
                    int em = tbase + m;
                    if (em < E) {
                        float msg = Wf[nt][q] * acc[q];
                        atomicAdd(&agg[(size_t)itile[wave][m] * FD + nt * 16 + lm], msg);
                    }
                }
            }
        }
    }
}

// ---------- forward node kernel ----------
__global__ void node_fwd_kernel(const float* h_in, const float* agg, const float* lin2w,
                                const float* lin2b, float* h_out, int N) {
    int idx = blockIdx.x * blockDim.x + threadIdx.x;
    if (idx >= N * HD) return;
    int n = idx / HD, c = idx % HD;
    float t = lin2b[c];
    const float* a = agg + (size_t)n * FD;
    for (int f = 0; f < FD; ++f) t += a[f] * lin2w[f * HD + c];
    h_out[idx] = h_in[idx] + sspf(t);
}

// ---------- output head: energy + dE/dh3 ----------
__global__ void head_kernel(const float* h, const int* batch, const float* w1,
                            const float* b1, const float* w2, const float* b2,
                            float* energy, float* dh, int N) {
    __shared__ float sq[64];
    __shared__ float ered[128];
    int n = blockIdx.x, t = threadIdx.x;
    float epart = 0.f;
    if (t < 64) {
        float t4 = b1[t];
        const float* hr = h + (size_t)n * HD;
        for (int c = 0; c < HD; ++c) t4 += hr[c] * w1[c * 64 + t];
        float w = w2[t];
        sq[t] = sigf(t4) * w;
        epart = sspf(t4) * w;
    }
    ered[t] = epart;
    __syncthreads();
    for (int s = 64; s > 0; s >>= 1) {
        if (t < s) ered[t] += ered[t + s];
        __syncthreads();
    }
    if (t == 0) atomicAdd(&energy[batch[n]], ered[0] + b2[0]);
    float acc = 0.f;
    for (int q = 0; q < 64; ++q) acc += sq[q] * w1[t * 64 + q];
    dh[(size_t)n * HD + t] = acc;
}

// ---------- backward node kernels ----------
__global__ void s3_kernel(const float* dh, const float* agg, const float* lin2w,
                          const float* lin2b, float* s3, int N) {
    int idx = blockIdx.x * blockDim.x + threadIdx.x;
    if (idx >= N * HD) return;
    int n = idx / HD, c = idx % HD;
    float t = lin2b[c];
    const float* a = agg + (size_t)n * FD;
    for (int f = 0; f < FD; ++f) t += a[f] * lin2w[f * HD + c];
    s3[idx] = dh[idx] * sigf(t);
}

__global__ void dagg_kernel(const float* s3, const float* lin2w, float* dagg, int N) {
    int idx = blockIdx.x * blockDim.x + threadIdx.x;
    if (idx >= N * FD) return;
    int n = idx / FD, f = idx % FD;
    float t = 0.f;
    const float* s = s3 + (size_t)n * HD;
    const float* w = lin2w + f * HD;
    for (int c = 0; c < HD; ++c) t += s[c] * w[c];
    dagg[idx] = t;
}

// ---------- backward edge kernel (WMMA) ----------
__global__ void edge_bwd_kernel(const int* ei, const float* dvec, const float* h_in,
                                const float* dagg, float* dh, float* dd,
                                const unsigned short* wb, const float* b1,
                                const float* b2, int E, int tiles) {
    __shared__ __align__(16) unsigned short rbfs[WPB][16 * 64];
    __shared__ __align__(16) unsigned short a1s[WPB][16 * 64];
    __shared__ __align__(16) unsigned short s1s[WPB][16 * 64];
    __shared__ __align__(16) unsigned short dgs[WPB][16 * 64];
    __shared__ __align__(16) unsigned short dt2s[WPB][16 * 64];
    __shared__ __align__(16) unsigned short dt1s[WPB][16 * 64];
    __shared__ float dtile[WPB][16];
    __shared__ float Ctile[WPB][16];
    __shared__ int   itile[WPB][16];
    __shared__ int   jtile[WPB][16];

    int wave = threadIdx.x >> 5, lane = threadIdx.x & 31;
    int lm = lane & 15, half = lane >> 4;
    int tile = blockIdx.x * WPB + wave;
    bool valid = tile < tiles;
    int tbase = valid ? tile * 16 : 0;

    if (lane < 16) {
        int e = tbase + lane; if (e >= E) e = E - 1;
        float d = dvec[e];
        dtile[wave][lane] = d;
        Ctile[wave][lane] = (d <= CUT) ? 0.5f * (cosf(d * PI_F / CUT) + 1.f) : 0.f;
        itile[wave][lane] = ei[e];
        jtile[wave][lane] = ei[E + e];
    }
    __syncthreads();

    for (int idx = lane; idx < 16 * 64; idx += 32) {
        int m = idx >> 6, k = idx & 63;
        float val = 0.f;
        if (k < NGAUSS) {
            float t = dtile[wave][m] - (CUT / (NGAUSS - 1)) * (float)k;
            val = expf(-50.f * t * t);
        }
        rbfs[wave][idx] = bfbits(val);
    }
    __syncthreads();

    const unsigned short* B1F = wb;
    const unsigned short* B2F = wb + 4096;
    const unsigned short* B3F = wb + 8192;
    const unsigned short* B2B = wb + 16384;
    const unsigned short* B1B = wb + 20480;
    const unsigned short* B3B = wb + 24576;

    // recompute t1 = rbf@W1+b1 ; a1=ssp, s1=sigmoid -> LDS
    {
        v16bf A0 = loadA_lds(rbfs[wave], 64, lm, half, 0);
        v16bf A1 = loadA_lds(rbfs[wave], 64, lm, half, 32);
#pragma unroll
        for (int nt = 0; nt < 4; ++nt) {
            float bias = b1[nt * 16 + lm];
            v8f acc;
#pragma unroll
            for (int q = 0; q < 8; ++q) acc[q] = bias;
            acc = wmma_bf16(A0, loadB(B1F, 64, nt * 16 + lm, half, 0), acc);
            acc = wmma_bf16(A1, loadB(B1F, 64, nt * 16 + lm, half, 32), acc);
#pragma unroll
            for (int q = 0; q < 8; ++q) {
                int m = q + 8 * half;
                a1s[wave][m * 64 + nt * 16 + lm] = bfbits(sspf(acc[q]));
                s1s[wave][m * 64 + nt * 16 + lm] = bfbits(sigf(acc[q]));
            }
        }
    }
    __syncthreads();

    // recompute t2 = a1@W2+b2 (kept in regs)
    v8f t2a[4];
    {
        v16bf A0 = loadA_lds(a1s[wave], 64, lm, half, 0);
        v16bf A1 = loadA_lds(a1s[wave], 64, lm, half, 32);
#pragma unroll
        for (int nt = 0; nt < 4; ++nt) {
            float bias = b2[nt * 16 + lm];
            v8f acc;
#pragma unroll
            for (int q = 0; q < 8; ++q) acc[q] = bias;
            acc = wmma_bf16(A0, loadB(B2F, 64, nt * 16 + lm, half, 0), acc);
            acc = wmma_bf16(A1, loadB(B2F, 64, nt * 16 + lm, half, 32), acc);
            t2a[nt] = acc;
        }
    }

    // recompute g = h[j]@lin1_w (kept in regs)
    v8f ga[4];
    {
        const float* hrow = h_in + (size_t)jtile[wave][lm] * HD;
        v16bf HA[4];
#pragma unroll
        for (int kk = 0; kk < 4; ++kk) HA[kk] = loadA_rowf(hrow, kk * 32, half);
#pragma unroll
        for (int nt = 0; nt < 4; ++nt) {
            v8f acc;
#pragma unroll
            for (int q = 0; q < 8; ++q) acc[q] = 0.f;
#pragma unroll
            for (int kk = 0; kk < 4; ++kk)
                acc = wmma_bf16(HA[kk], loadB(B3F, 128, nt * 16 + lm, half, kk * 32), acc);
            ga[nt] = acc;
        }
    }

    // dmsg = dagg[i]; dWf = dmsg*g; dg = dmsg*Wf; dt2 = dWf*C; pdC += dWf*t2
    float pdC[8];
#pragma unroll
    for (int q = 0; q < 8; ++q) pdC[q] = 0.f;
#pragma unroll
    for (int nt = 0; nt < 4; ++nt) {
#pragma unroll
        for (int q = 0; q < 8; ++q) {
            int m = q + 8 * half;
            float Cm = Ctile[wave][m];
            float dmsg = dagg[(size_t)itile[wave][m] * FD + nt * 16 + lm];
            float dWf = dmsg * ga[nt][q];
            float dg  = dmsg * t2a[nt][q] * Cm;     // dmsg * Wf
            pdC[q] += dWf * t2a[nt][q];
            dgs[wave][m * 64 + nt * 16 + lm]  = bfbits(dg);
            dt2s[wave][m * 64 + nt * 16 + lm] = bfbits(dWf * Cm);
        }
    }
    __syncthreads();

    // reduce pdC over the 16 n-lanes within each half
#pragma unroll
    for (int q = 0; q < 8; ++q) {
        float v = pdC[q];
        v += __shfl_xor(v, 1); v += __shfl_xor(v, 2);
        v += __shfl_xor(v, 4); v += __shfl_xor(v, 8);
        pdC[q] = v;
    }

    // da1 = dt2 @ W2^T ; dt1 = da1 * s1 -> LDS
    {
        v16bf A0 = loadA_lds(dt2s[wave], 64, lm, half, 0);
        v16bf A1 = loadA_lds(dt2s[wave], 64, lm, half, 32);
#pragma unroll
        for (int nt = 0; nt < 4; ++nt) {
            v8f acc;
#pragma unroll
            for (int q = 0; q < 8; ++q) acc[q] = 0.f;
            acc = wmma_bf16(A0, loadB(B2B, 64, nt * 16 + lm, half, 0), acc);
            acc = wmma_bf16(A1, loadB(B2B, 64, nt * 16 + lm, half, 32), acc);
#pragma unroll
            for (int q = 0; q < 8; ++q) {
                int m = q + 8 * half;
                float s1 = (float)bffrom(s1s[wave][m * 64 + nt * 16 + lm]);
                dt1s[wave][m * 64 + nt * 16 + lm] = bfbits(acc[q] * s1);
            }
        }
    }
    __syncthreads();

    // drbf = dt1 @ W1^T ; fold into dd via d(rbf)/dd (rbf read back from LDS)
    float prb[8];
#pragma unroll
    for (int q = 0; q < 8; ++q) prb[q] = 0.f;
    {
        v16bf A0 = loadA_lds(dt1s[wave], 64, lm, half, 0);
        v16bf A1 = loadA_lds(dt1s[wave], 64, lm, half, 32);
#pragma unroll
        for (int nt = 0; nt < 4; ++nt) {
            v8f acc;
#pragma unroll
            for (int q = 0; q < 8; ++q) acc[q] = 0.f;
            acc = wmma_bf16(A0, loadB(B1B, 64, nt * 16 + lm, half, 0), acc);
            acc = wmma_bf16(A1, loadB(B1B, 64, nt * 16 + lm, half, 32), acc);
            int k = nt * 16 + lm;
            if (k < NGAUSS) {
#pragma unroll
                for (int q = 0; q < 8; ++q) {
                    int m = q + 8 * half;
                    float dmq = dtile[wave][m];
                    float t = dmq - (CUT / (NGAUSS - 1)) * (float)k;
                    float rbf = (float)bffrom(rbfs[wave][m * 64 + k]);
                    // d rbf/dd = rbf * 2*coeff*(d-off) = rbf * (-100)*(d-off)
                    prb[q] += acc[q] * rbf * (-100.f) * t;
                }
            }
        }
    }
#pragma unroll
    for (int q = 0; q < 8; ++q) {
        float v = prb[q];
        v += __shfl_xor(v, 1); v += __shfl_xor(v, 2);
        v += __shfl_xor(v, 4); v += __shfl_xor(v, 8);
        prb[q] = v;
    }
    if (valid && lm == 0) {
#pragma unroll
        for (int q = 0; q < 8; ++q) {
            int m = q + 8 * half;
            int em = tbase + m;
            if (em < E) {
                float dmq = dtile[wave][m];
                float cp = (dmq <= CUT) ? -0.5f * sinf(dmq * PI_F / CUT) * (PI_F / CUT) : 0.f;
                dd[em] += prb[q] + pdC[q] * cp;
            }
        }
    }

    // dh[j] += dg @ lin1_w^T  (8 n-tiles over H=128)
    {
        v16bf A0 = loadA_lds(dgs[wave], 64, lm, half, 0);
        v16bf A1 = loadA_lds(dgs[wave], 64, lm, half, 32);
        for (int nt = 0; nt < 8; ++nt) {
            v8f acc;
#pragma unroll
            for (int q = 0; q < 8; ++q) acc[q] = 0.f;
            acc = wmma_bf16(A0, loadB(B3B, 64, nt * 16 + lm, half, 0), acc);
            acc = wmma_bf16(A1, loadB(B3B, 64, nt * 16 + lm, half, 32), acc);
            if (valid) {
#pragma unroll
                for (int q = 0; q < 8; ++q) {
                    int m = q + 8 * half;
                    int em = tbase + m;
                    if (em < E)
                        atomicAdd(&dh[(size_t)jtile[wave][m] * HD + nt * 16 + lm], acc[q]);
                }
            }
        }
    }
}

// ---------- forces ----------
__global__ void force_kernel(const float* ew, const int* ei, const float* dvec,
                             const float* dd, float* forces, int E) {
    int e = blockIdx.x * blockDim.x + threadIdx.x;
    if (e >= E) return;
    float d = dvec[e];
    float s = dd[e] / fmaxf(d, 1e-12f);
    float fx = s * ew[3 * e], fy = s * ew[3 * e + 1], fz = s * ew[3 * e + 2];
    int i = ei[e], j = ei[E + e];
    atomicAdd(&forces[3 * i + 0], fx);
    atomicAdd(&forces[3 * i + 1], fy);
    atomicAdd(&forces[3 * i + 2], fz);
    atomicAdd(&forces[3 * j + 0], -fx);
    atomicAdd(&forces[3 * j + 1], -fy);
    atomicAdd(&forces[3 * j + 2], -fz);
}

// ---------- host launcher ----------
extern "C" void kernel_launch(void* const* d_in, const int* in_sizes, int n_in,
                              void* d_out, int out_size, void* d_ws, size_t ws_size,
                              hipStream_t stream) {
    (void)n_in; (void)ws_size;
    // ---- determine pytree flattening order from leaf sizes ----
    int ix_x = 0, ix_ei = 1, ix_ew = 2, ix_batch = 3, pb;
    if (in_sizes[3] == 12800) {          // fully-sorted tree: batch,ei,ew,embed,...,x
        ix_batch = 0; ix_ei = 1; ix_ew = 2; pb = 3; ix_x = 29;
    } else {                             // top-level insertion order: x,ei,ew,batch,params...
        pb = 4;
    }
    int ix_embed = pb;
    int lbase = pb + 1;
    bool layerSorted = (in_sizes[lbase] == 8192);   // lin1_w first (sorted) vs mlp_w1 (3200)
    int o_lin1, o_l2b, o_l2w, o_b1, o_b2, o_w1, o_w2;
    if (layerSorted) { o_lin1 = 0; o_l2b = 1; o_l2w = 2; o_b1 = 3; o_b2 = 4; o_w1 = 5; o_w2 = 6; }
    else             { o_w1 = 0; o_b1 = 1; o_w2 = 2; o_b2 = 3; o_lin1 = 4; o_l2w = 5; o_l2b = 6; }
    int ob = pb + 22;                    // output-head block
    int ix_ow1, ix_ob1, ix_ow2, ix_ob2;
    if (in_sizes[ob] == 8192) { ix_ow1 = ob; ix_ob1 = ob + 1; ix_ow2 = ob + 2; ix_ob2 = ob + 3; }
    else                      { ix_ob1 = ob; ix_ob2 = ob + 1; ix_ow1 = ob + 2; ix_ow2 = ob + 3; }

    const int N = in_sizes[ix_x];
    const int E = in_sizes[ix_ei] / 2;
    const int G = out_size - 3 * N;

    const int*   x     = (const int*)d_in[ix_x];
    const int*   ei    = (const int*)d_in[ix_ei];
    const float* ew    = (const float*)d_in[ix_ew];
    const int*   batch = (const int*)d_in[ix_batch];
    const float* emb   = (const float*)d_in[ix_embed];
    const float* lw1[3], *lb1[3], *lw2[3], *lb2[3], *lin1[3], *lin2w[3], *lin2b[3];
    for (int l = 0; l < 3; ++l) {
        int b = lbase + 7 * l;
        lw1[l]   = (const float*)d_in[b + o_w1];
        lb1[l]   = (const float*)d_in[b + o_b1];
        lw2[l]   = (const float*)d_in[b + o_w2];
        lb2[l]   = (const float*)d_in[b + o_b2];
        lin1[l]  = (const float*)d_in[b + o_lin1];
        lin2w[l] = (const float*)d_in[b + o_l2w];
        lin2b[l] = (const float*)d_in[b + o_l2b];
    }
    const float* ow1 = (const float*)d_in[ix_ow1];
    const float* ob1 = (const float*)d_in[ix_ob1];
    const float* ow2 = (const float*)d_in[ix_ow2];
    const float* ob2 = (const float*)d_in[ix_ob2];

    // ---- workspace layout (floats) ----
    float* ws = (float*)d_ws;
    float* dvec = ws;                       size_t off = (size_t)E;
    float* dd   = ws + off;                 off += (size_t)E;
    float* h[4];
    for (int l = 0; l < 4; ++l) { h[l] = ws + off; off += (size_t)N * HD; }
    float* agg[3];
    for (int l = 0; l < 3; ++l) { agg[l] = ws + off; off += (size_t)N * FD; }
    float* dh   = ws + off;                 off += (size_t)N * HD;
    float* dagg = ws + off;                 off += (size_t)N * FD;
    float* s3   = ws + off;                 off += (size_t)N * HD;
    unsigned short* wshort = (unsigned short*)(ws + off);

    float* energy = (float*)d_out;          // [G]
    float* forces = (float*)d_out + G;      // [N,3]

    const int tiles = (E + 15) / 16;
    const int eblocks = (tiles + WPB - 1) / WPB;

    // bf16 weight tables
    for (int l = 0; l < 3; ++l)
        prep_kernel<<<(LAYER_WSHORTS + 255) / 256, 256, 0, stream>>>(
            lw1[l], lw2[l], lin1[l], wshort + (size_t)l * LAYER_WSHORTS);

    // zero outputs + scatter targets
    zero_kernel<<<256, 256, 0, stream>>>((float*)d_out, (long)out_size);
    zero_kernel<<<256, 256, 0, stream>>>(agg[0], (long)3 * N * FD);

    geom_kernel<<<(E + 255) / 256, 256, 0, stream>>>(ew, dvec, dd, E);
    embed_kernel<<<(N * HD + 255) / 256, 256, 0, stream>>>(x, emb, h[0], N);

    // forward
    for (int l = 0; l < 3; ++l) {
        edge_fwd_kernel<<<eblocks, 32 * WPB, 0, stream>>>(
            ei, dvec, h[l], agg[l], wshort + (size_t)l * LAYER_WSHORTS,
            lb1[l], lb2[l], E, tiles);
        node_fwd_kernel<<<(N * HD + 255) / 256, 256, 0, stream>>>(
            h[l], agg[l], lin2w[l], lin2b[l], h[l + 1], N);
    }

    // output head: energy + dE/dh3
    head_kernel<<<N, 128, 0, stream>>>(h[3], batch, ow1, ob1, ow2, ob2, energy, dh, N);

    // backward
    for (int l = 2; l >= 0; --l) {
        s3_kernel<<<(N * HD + 255) / 256, 256, 0, stream>>>(
            dh, agg[l], lin2w[l], lin2b[l], s3, N);
        dagg_kernel<<<(N * FD + 255) / 256, 256, 0, stream>>>(s3, lin2w[l], dagg, N);
        edge_bwd_kernel<<<eblocks, 32 * WPB, 0, stream>>>(
            ei, dvec, h[l], dagg, dh, dd, wshort + (size_t)l * LAYER_WSHORTS,
            lb1[l], lb2[l], E, tiles);
    }

    force_kernel<<<(E + 255) / 256, 256, 0, stream>>>(ew, ei, dvec, dd, forces, E);
}
